// SarvamMoEAttention_68410239091012
// MI455X (gfx1250) — matrix operand order repaired and checked by reading, not compiled, not run on hardware
//
#include <hip/hip_runtime.h>
#include <hip/hip_bf16.h>

// ---------------------------------------------------------------------------
// Types for CDNA5 WMMA (wave32): 16x16x32 bf16 -> f32
// ---------------------------------------------------------------------------
typedef __attribute__((ext_vector_type(16))) __bf16 v16bf;
typedef __attribute__((ext_vector_type(8)))  __bf16 v8bf;
typedef __attribute__((ext_vector_type(8)))  float  v8f;

union BF16x16 { v16bf v; v8bf h[2]; };

__device__ __forceinline__ v8f wmma_bf16(const v16bf& a, const v16bf& b, const v8f& c) {
    return __builtin_amdgcn_wmma_f32_16x16x32_bf16(
        /*neg_a=*/false, a, /*neg_b=*/false, b,
        /*c_mod=*/(short)0, c, /*reuse_a=*/false, /*reuse_b=*/false);
}

__device__ __forceinline__ float redmax16(float v) {
    v = fmaxf(v, __shfl_xor(v, 1));
    v = fmaxf(v, __shfl_xor(v, 2));
    v = fmaxf(v, __shfl_xor(v, 4));
    v = fmaxf(v, __shfl_xor(v, 8));
    return v;
}
__device__ __forceinline__ float redsum16(float v) {
    v += __shfl_xor(v, 1);
    v += __shfl_xor(v, 2);
    v += __shfl_xor(v, 4);
    v += __shfl_xor(v, 8);
    return v;
}
__device__ __forceinline__ float redsum32(float v) {
    v += __shfl_xor(v, 1);
    v += __shfl_xor(v, 2);
    v += __shfl_xor(v, 4);
    v += __shfl_xor(v, 8);
    v += __shfl_xor(v, 16);
    return v;
}

// ---------------------------------------------------------------------------
// Kernel 1/4: C[MxN] = A[MxK] * B[KxN]; fp32 in memory, bf16 WMMA compute.
// Block tile 128x128, k-step 64, 256 threads = 8 waves (4 M-waves x 2 N-waves),
// each wave computes a 32x64 sub-tile = 2x4 WMMA accumulators.
// Double-buffered LDS: tile i+1 is register-staged (16x global_load_b128)
// while the 16 WMMAs of tile i run; one barrier per k-tile.
// ---------------------------------------------------------------------------
#define GT_M 128
#define GT_N 128
#define GT_K 64
#define LDSP 72   // padded row length (bf16 elems): 16B-aligned, breaks conflicts

__global__ __launch_bounds__(256)
void gemm_f32_bf16w(const float* __restrict__ A, const float* __restrict__ B,
                    float* __restrict__ C, int M, int N, int K)
{
    __shared__ __bf16 lA[2][GT_M * LDSP];   // [buf][m][k]
    __shared__ __bf16 lB[2][GT_N * LDSP];   // [buf][n][k]  (transposed on store)

    const int tid  = threadIdx.x;
    const int lane = tid & 31;
    const int wave = tid >> 5;
    const int wm   = wave & 3;   // m offset wm*32
    const int wn   = wave >> 2;  // n offset wn*64
    const int mb   = blockIdx.y * GT_M;
    const int nb   = blockIdx.x * GT_N;
    const int ln   = lane & 15;
    const int kg   = lane >> 4;

    float4 ra[8], rb[8];   // register-staged next tile

    auto load_regs = [&](int k0) {
        #pragma unroll
        for (int i = 0; i < 8; ++i) {        // A tile: 128 x 64, 16 float4/row
            int f = tid + i * 256;
            ra[i] = *(const float4*)(A + (size_t)(mb + (f >> 4)) * K + k0 + (f & 15) * 4);
        }
        #pragma unroll
        for (int i = 0; i < 8; ++i) {        // B tile: 64 x 128, 32 float4/row
            int f = tid + i * 256;
            rb[i] = *(const float4*)(B + (size_t)(k0 + (f >> 5)) * N + nb + (f & 31) * 4);
        }
    };
    auto store_lds = [&](int buf) {
        #pragma unroll
        for (int i = 0; i < 8; ++i) {
            int f = tid + i * 256;
            __bf16* dst = &lA[buf][(f >> 4) * LDSP + (f & 15) * 4];
            dst[0] = (__bf16)ra[i].x; dst[1] = (__bf16)ra[i].y;
            dst[2] = (__bf16)ra[i].z; dst[3] = (__bf16)ra[i].w;
        }
        #pragma unroll
        for (int i = 0; i < 8; ++i) {
            int f  = tid + i * 256;
            int kr = f >> 5, c4 = f & 31;
            lB[buf][(c4 * 4 + 0) * LDSP + kr] = (__bf16)rb[i].x;
            lB[buf][(c4 * 4 + 1) * LDSP + kr] = (__bf16)rb[i].y;
            lB[buf][(c4 * 4 + 2) * LDSP + kr] = (__bf16)rb[i].z;
            lB[buf][(c4 * 4 + 3) * LDSP + kr] = (__bf16)rb[i].w;
        }
    };

    v8f acc[2][4];
    for (int mt = 0; mt < 2; ++mt)
        for (int nt = 0; nt < 4; ++nt)
            for (int j = 0; j < 8; ++j) acc[mt][nt][j] = 0.0f;

    const int ntiles = K / GT_K;
    load_regs(0);
    store_lds(0);
    int cur = 0;

    for (int it = 0; it < ntiles; ++it) {
        __syncthreads();                       // buf[cur] visible to all waves
        const bool more = (it + 1 < ntiles);
        if (more) load_regs((it + 1) * GT_K);  // issue loads; no wait yet

        #pragma unroll
        for (int kc = 0; kc < 2; ++kc) {       // two 32-deep WMMA chunks
            BF16x16 af[2];
            #pragma unroll
            for (int mt = 0; mt < 2; ++mt) {
                int m = wm * 32 + mt * 16 + ln;
                const __bf16* p = &lA[cur][m * LDSP + kc * 32 + kg * 16];
                af[mt].h[0] = *(const v8bf*)(p);        // K = kc*32 + kg*16 + [0..7]
                af[mt].h[1] = *(const v8bf*)(p + 8);    //            ... + [8..15]
            }
            BF16x16 bfm[4];
            #pragma unroll
            for (int nt = 0; nt < 4; ++nt) {
                int n = wn * 64 + nt * 16 + ln;
                const __bf16* p = &lB[cur][n * LDSP + kc * 32 + kg * 16];
                bfm[nt].h[0] = *(const v8bf*)(p);       // K = kc*32 + kg*16 + [0..7]
                bfm[nt].h[1] = *(const v8bf*)(p + 8);   //            ... + [8..15]
            }
            #pragma unroll
            for (int mt = 0; mt < 2; ++mt)
                #pragma unroll
                for (int nt = 0; nt < 4; ++nt)
                    acc[mt][nt] = wmma_bf16(af[mt].v, bfm[nt].v, acc[mt][nt]);
        }

        if (more) store_lds(cur ^ 1);          // waits loads here, overlapped above
        cur ^= 1;
    }

    // epilogue: C/D layout -> lane ln = column, VGPR j -> row j + kg*8
    #pragma unroll
    for (int mt = 0; mt < 2; ++mt)
        #pragma unroll
        for (int nt = 0; nt < 4; ++nt)
            #pragma unroll
            for (int j = 0; j < 8; ++j) {
                int m = mb + wm * 32 + mt * 16 + j + kg * 8;
                int n = nb + wn * 64 + nt * 16 + ln;
                C[(size_t)m * N + n] = acc[mt][nt][j];
            }
}

// ---------------------------------------------------------------------------
// Kernel 2: per-token RMSNorm + RoPE + pack to bf16 (Q and K only).
//   q  -> [32][T][128] row-major  (WMMA A layout source)
//   k  -> [8][T][128]  row-major  (== WMMA B layout for QK^T)
// ---------------------------------------------------------------------------
__global__ __launch_bounds__(256)
void norm_rope(const int* __restrict__ positions, const float* __restrict__ qkv,
               const float* __restrict__ qw, const float* __restrict__ kw,
               __bf16* __restrict__ qb, __bf16* __restrict__ kb, int T)
{
    const int t    = blockIdx.x;
    const int lane = threadIdx.x & 31;
    const int wave = threadIdx.x >> 5;
    const float pos = (float)positions[t];

    for (int r = wave; r < 40; r += 8) {      // 32 q rows + 8 k rows
        const float* src = qkv + (size_t)t * 6144 + r * 128;
        float x0 = src[lane], x1 = src[lane + 32], x2 = src[lane + 64], x3 = src[lane + 96];

        float ss = redsum32(x0 * x0 + x1 * x1 + x2 * x2 + x3 * x3);
        float rinv = rsqrtf(ss * (1.0f / 128.0f) + 1e-6f);
        const float* w = (r < 32) ? qw : kw;
        x0 *= rinv * w[lane];      x1 *= rinv * w[lane + 32];
        x2 *= rinv * w[lane + 64]; x3 *= rinv * w[lane + 96];

        // RoPE pairs: (d, d+64). lane covers d = lane and d = lane+32.
        float f0 = __powf(10000.0f, -(float)lane        * (1.0f / 64.0f));
        float f1 = __powf(10000.0f, -(float)(lane + 32) * (1.0f / 64.0f));
        float a0 = pos * f0, a1 = pos * f1;
        float c0 = __cosf(a0), s0 = __sinf(a0);
        float c1 = __cosf(a1), s1 = __sinf(a1);
        float y0 = x0 * c0 - x2 * s0;
        float y2 = x2 * c0 + x0 * s0;
        float y1 = x1 * c1 - x3 * s1;
        float y3 = x3 * c1 + x1 * s1;

        __bf16* dst = (r < 32) ? (qb + ((size_t)r * T + t) * 128)
                               : (kb + ((size_t)(r - 32) * T + t) * 128);
        dst[lane]      = (__bf16)y0;
        dst[lane + 32] = (__bf16)y1;
        dst[lane + 64] = (__bf16)y2;
        dst[lane + 96] = (__bf16)y3;
    }
}

// ---------------------------------------------------------------------------
// Kernel 3: V transpose through LDS: qkv v-section (fp32, [t][d]) ->
//   vt bf16 [8][128][T]  (== WMMA B layout for P.V: contiguous contraction idx)
// Coalesced float4 reads, coalesced 64B row writes; LDS tile [d][t] padded.
// ---------------------------------------------------------------------------
__global__ __launch_bounds__(256)
void v_transpose(const float* __restrict__ qkv, __bf16* __restrict__ vt, int T)
{
    __shared__ __bf16 tile[128 * 72];   // [d][t0..t0+63], 72-elem rows (16B align)
    const int tid = threadIdx.x;
    const int t0  = blockIdx.x * 64;
    const int h   = blockIdx.y;

    #pragma unroll
    for (int i = 0; i < 8; ++i) {
        int f  = tid + i * 256;          // 2048 float4 slots = 64 t x 32 float4
        int tt = f >> 5;
        int c4 = f & 31;
        float4 d = *(const float4*)(qkv + (size_t)(t0 + tt) * 6144 + 5120 + h * 128 + c4 * 4);
        tile[(c4 * 4 + 0) * 72 + tt] = (__bf16)d.x;
        tile[(c4 * 4 + 1) * 72 + tt] = (__bf16)d.y;
        tile[(c4 * 4 + 2) * 72 + tt] = (__bf16)d.z;
        tile[(c4 * 4 + 3) * 72 + tt] = (__bf16)d.w;
    }
    __syncthreads();

    const int d = tid >> 1, half = tid & 1;  // 2 threads per d-row, 32 t each
    const __bf16* src = &tile[d * 72 + half * 32];
    __bf16* dst = vt + ((size_t)h * 128 + d) * T + t0 + half * 32;
    #pragma unroll
    for (int i = 0; i < 4; ++i)
        ((v8bf*)dst)[i] = ((const v8bf*)src)[i];
}

// ---------------------------------------------------------------------------
// Kernel 4: causal GQA flash attention. One wave per (16-row q-tile, head).
// k-tiles of 32 columns; online softmax; P transposed C->A layout via LDS.
// ---------------------------------------------------------------------------
__global__ __launch_bounds__(32)
void attn(const __bf16* __restrict__ qb, const __bf16* __restrict__ kb,
          const __bf16* __restrict__ vt, float* __restrict__ out, int T)
{
    __shared__ __bf16 lp[16 * 32];           // P tile, row-major [m][k]

    const int lane = threadIdx.x;
    const int ln   = lane & 15;
    const int kg   = lane >> 4;
    const int qt   = blockIdx.x;
    const int h    = blockIdx.y;
    const int hk   = h >> 2;                 // GQA: 4 q heads per kv head
    const int q0   = qt * 16;

    const __bf16* qh = qb + (size_t)h  * T * 128;
    const __bf16* kh = kb + (size_t)hk * T * 128;
    const __bf16* vh = vt + (size_t)hk * 128 * T;

    // Q A-fragments (16 rows x 128 d, 4 chunks of 32)
    BF16x16 qa[4];
    #pragma unroll
    for (int c = 0; c < 4; ++c) {
        const __bf16* p = qh + (size_t)(q0 + ln) * 128 + c * 32 + kg * 16;
        qa[c].h[0] = *(const v8bf*)(p);
        qa[c].h[1] = *(const v8bf*)(p + 8);
    }

    v8f o[8];
    #pragma unroll
    for (int nt = 0; nt < 8; ++nt)
        for (int j = 0; j < 8; ++j) o[nt][j] = 0.0f;
    float mrow[8], lrow[8];
    #pragma unroll
    for (int j = 0; j < 8; ++j) { mrow[j] = -3.0e38f; lrow[j] = 0.0f; }

    const float scale = 0.08838834764831845f;   // 1/sqrt(128)
    const int kend = q0 + 16;                   // causal: need k <= q0+15

    for (int kt = 0; kt < kend; kt += 32) {
        // ---- S = Q K^T : two 16x16 accumulators over 32 k-columns
        v8f s0 = {0,0,0,0,0,0,0,0}, s1 = {0,0,0,0,0,0,0,0};
        #pragma unroll
        for (int c = 0; c < 4; ++c) {
            BF16x16 b0, b1;
            const __bf16* p0 = kh + (size_t)(kt + ln) * 128 + c * 32 + kg * 16;
            b0.h[0] = *(const v8bf*)(p0);
            b0.h[1] = *(const v8bf*)(p0 + 8);
            const __bf16* p1 = kh + (size_t)(kt + 16 + ln) * 128 + c * 32 + kg * 16;
            b1.h[0] = *(const v8bf*)(p1);
            b1.h[1] = *(const v8bf*)(p1 + 8);
            s0 = wmma_bf16(qa[c].v, b0.v, s0);
            s1 = wmma_bf16(qa[c].v, b1.v, s1);
        }
        // ---- mask + online softmax per row (row m = j + kg*8)
        #pragma unroll
        for (int j = 0; j < 8; ++j) {
            int qrow = q0 + j + kg * 8;
            float x0 = s0[j] * scale; if (kt + ln      > qrow) x0 = -1.0e30f;
            float x1 = s1[j] * scale; if (kt + 16 + ln > qrow) x1 = -1.0e30f;
            float rm = redmax16(fmaxf(x0, x1));
            float mn = fmaxf(mrow[j], rm);
            float corr = __expf(mrow[j] - mn);
            float p0 = __expf(x0 - mn);
            float p1 = __expf(x1 - mn);
            float rs = redsum16(p0 + p1);
            lrow[j] = lrow[j] * corr + rs;
            mrow[j] = mn;
            #pragma unroll
            for (int nt = 0; nt < 8; ++nt) o[nt][j] *= corr;
            int m = j + kg * 8;
            lp[m * 32 + ln]      = (__bf16)p0;
            lp[m * 32 + 16 + ln] = (__bf16)p1;
        }
        asm volatile("s_wait_dscnt 0" ::: "memory");   // cross-lane LDS RAW

        // ---- P (A-layout from LDS) x V^T (B-layout from global)
        BF16x16 pa;
        {
            const __bf16* p = &lp[ln * 32 + kg * 16];
            pa.h[0] = *(const v8bf*)(p);
            pa.h[1] = *(const v8bf*)(p + 8);
        }
        #pragma unroll
        for (int nt = 0; nt < 8; ++nt) {
            BF16x16 vb;
            const __bf16* p = vh + (size_t)(nt * 16 + ln) * T + kt + kg * 16;
            vb.h[0] = *(const v8bf*)(p);
            vb.h[1] = *(const v8bf*)(p + 8);
            o[nt] = wmma_bf16(pa.v, vb.v, o[nt]);
        }
    }

    // ---- normalize and store fp32 into attn buffer [T][4096]
    #pragma unroll
    for (int j = 0; j < 8; ++j) {
        float inv = 1.0f / lrow[j];
        int row = q0 + j + kg * 8;
        #pragma unroll
        for (int nt = 0; nt < 8; ++nt)
            out[(size_t)row * 4096 + h * 128 + nt * 16 + ln] = o[nt][j] * inv;
    }
}

// ---------------------------------------------------------------------------
// Host launch
// ---------------------------------------------------------------------------
extern "C" void kernel_launch(void* const* d_in, const int* in_sizes, int n_in,
                              void* d_out, int out_size, void* d_ws, size_t ws_size,
                              hipStream_t stream)
{
    const int*   positions = (const int*)d_in[0];
    const float* hidden    = (const float*)d_in[1];
    const float* w_qkv     = (const float*)d_in[2];
    const float* q_norm_w  = (const float*)d_in[3];
    const float* k_norm_w  = (const float*)d_in[4];
    const float* w_o       = (const float*)d_in[5];
    float*       out       = (float*)d_out;

    const int T = 2048, H = 4096, NQKV = 6144;

    char* ws = (char*)d_ws;
    float*  qkv = (float*)ws;                               // 48 MiB
    size_t off  = (size_t)T * NQKV * sizeof(float);
    __bf16* qb  = (__bf16*)(ws + off); off += (size_t)32 * T * 128 * 2;  // 16 MiB
    __bf16* kb  = (__bf16*)(ws + off); off += (size_t)8  * T * 128 * 2;  //  4 MiB
    __bf16* vt  = (__bf16*)(ws + off); off += (size_t)8  * 128 * T * 2;  //  4 MiB
    float*  attn_buf = (float*)ws;   // aliases qkv region (dead after v_transpose)

    // 1) QKV projection: [T,H] x [H,6144]
    dim3 g1(NQKV / GT_N, T / GT_M);
    gemm_f32_bf16w<<<g1, 256, 0, stream>>>(hidden, w_qkv, qkv, T, NQKV, H);

    // 2) RMSNorm + RoPE + bf16 pack (Q, K)
    norm_rope<<<T, 256, 0, stream>>>(positions, qkv, q_norm_w, k_norm_w, qb, kb, T);

    // 3) V transpose to [hk][d][t] bf16
    dim3 g2(T / 64, 8);
    v_transpose<<<g2, 256, 0, stream>>>(qkv, vt, T);

    // 4) causal GQA attention -> attn_buf [T, 4096] fp32
    dim3 g3(T / 16, 32);
    attn<<<g3, 32, 0, stream>>>(qb, kb, vt, attn_buf, T);

    // 5) output projection: [T,4096] x [4096,4096]
    dim3 g4(H / GT_N, T / GT_M);
    gemm_f32_bf16w<<<g4, 256, 0, stream>>>(attn_buf, w_o, out, T, H, H);
}
